// FixedGraphSAGE_56066503082343
// MI455X (gfx1250) — matrix operand order repaired and checked by reading, not compiled, not run on hardware
//
#include <hip/hip_runtime.h>

#define NN 50000
#define NE 800000
#define IND 128
#define OUTD 64
#define BN_EPS 1e-5f

typedef __attribute__((ext_vector_type(16))) __bf16 v16bf;
typedef __attribute__((ext_vector_type(8)))  float  v8f;

__device__ __forceinline__ unsigned short f2bf(float f) {
  union { float f; unsigned int u; } v; v.f = f;
  unsigned int r = v.u + 0x7FFFu + ((v.u >> 16) & 1u);
  return (unsigned short)(r >> 16);
}
__device__ __forceinline__ __bf16 bfbits(unsigned short u) {
  union { unsigned short u; __bf16 b; } v; v.u = u; return v.b;
}

__global__ void fill_zero_i32(int* __restrict__ p, int n) {
  int i = blockIdx.x * blockDim.x + threadIdx.x;
  if (i < n) p[i] = 0;
}

__global__ void f32_to_bf16_k(const float* __restrict__ s, unsigned short* __restrict__ d, int n) {
  int i = blockIdx.x * blockDim.x + threadIdx.x;
  if (i < n) d[i] = f2bf(s[i]);
}

__global__ void degree_k(const int* __restrict__ dst, int* __restrict__ cnt, int e) {
  int i = blockIdx.x * blockDim.x + threadIdx.x;
  if (i < e) atomicAdd(&cnt[dst[i]], 1);
}

// Single-workgroup exclusive scan over n counts -> row offsets (n+1 entries).
// Runs once per launch on 50k elements; cost is negligible.
__global__ void exscan_k(const int* __restrict__ cnt, int* __restrict__ row, int n) {
  __shared__ int sdata[256];
  __shared__ int srun;
  if (threadIdx.x == 0) srun = 0;
  __syncthreads();
  for (int base = 0; base < n; base += 256) {
    int i = base + threadIdx.x;
    int v = (i < n) ? cnt[i] : 0;
    sdata[threadIdx.x] = v;
    __syncthreads();
#pragma unroll
    for (int off = 1; off < 256; off <<= 1) {
      int t = (threadIdx.x >= off) ? sdata[threadIdx.x - off] : 0;
      __syncthreads();
      sdata[threadIdx.x] += t;
      __syncthreads();
    }
    int excl = sdata[threadIdx.x] - v;
    if (i < n) row[i] = srun + excl;
    __syncthreads();
    if (threadIdx.x == 255) srun += sdata[255];
    __syncthreads();
  }
  if (threadIdx.x == 0) row[n] = srun;
}

// Bucket edges into CSR order (sorted by dst). ~E atomics, once per launch.
__global__ void place_k(const int* __restrict__ src, const int* __restrict__ dst,
                        const float* __restrict__ ew, const int* __restrict__ row,
                        int* __restrict__ cursor, int* __restrict__ esrc,
                        float* __restrict__ eww, int e) {
  int i = blockIdx.x * blockDim.x + threadIdx.x;
  if (i >= e) return;
  int d = dst[i];
  int pos = atomicAdd(&cursor[d], 1);
  int idx = row[d] + pos;
  esrc[idx] = src[i];
  eww[idx]  = ew[i];
}

// Fully fused SAGE layer:
//   per 16-node tile: gather-mean aggregation (CSR, f32 in VGPRs, no atomics)
//   -> bf16 LDS tiles -> Y = epilogue( mean @ Wl^T + b + X @ Wr^T ) via
//   v_wmma_f32_16x16x32_bf16. 256 threads = 8 waves; wave w owns output
//   features [16w,16w+16) and aggregates nodes {2w, 2w+1} of the tile.
__global__ __launch_bounds__(256) void sage_fused_k(
    const float* __restrict__ xin,
    const int* __restrict__ row,
    const int* __restrict__ esrc,
    const float* __restrict__ eww,
    const unsigned short* __restrict__ wl,
    const unsigned short* __restrict__ wr,
    const float* __restrict__ bias,
    const float* __restrict__ bng, const float* __restrict__ bnb,
    const float* __restrict__ bnm, const float* __restrict__ bnv,
    float* __restrict__ y,
    int dout, int apply_bn) {
  __shared__ unsigned short sM[16][IND + 8];   // aggregated means (bf16)
  __shared__ unsigned short sX[16][IND + 8];   // root activations (bf16)
  int tile = blockIdx.x;
  int tid  = threadIdx.x;
  int wv   = tid >> 5;
  int lane = tid & 31;

  // Phase 0: stage root activations for the tile (coalesced, all 256 threads).
  for (int idx = tid; idx < 16 * IND; idx += 256) {
    int nl = idx >> 7;
    int k  = idx & (IND - 1);
    sX[nl][k] = f2bf(xin[(size_t)(tile * 16 + nl) * IND + k]);
  }

  // Phase 1: gather-mean aggregation, 2 nodes per wave, f32 accumulation.
#pragma unroll
  for (int t = 0; t < 2; ++t) {
    int nl   = wv * 2 + t;
    int node = tile * 16 + nl;
    int beg  = row[node];
    int end  = row[node + 1];
    float4 acc = {0.f, 0.f, 0.f, 0.f};
    for (int j = beg; j < end; ++j) {
      int s   = esrc[j];
      float w = eww[j];
      if (j + 1 < end) {
        // gfx1250 global_prefetch_b8 hint for the next data-dependent row
        __builtin_prefetch(xin + (size_t)esrc[j + 1] * IND + lane * 4, 0, 0);
      }
      float4 v = ((const float4*)(xin + (size_t)s * IND))[lane];
      acc.x += v.x * w;
      acc.y += v.y * w;
      acc.z += v.z * w;
      acc.w += v.w * w;
    }
    float inv = 1.0f / fmaxf((float)(end - beg), 1.0f);
    sM[nl][lane * 4 + 0] = f2bf(acc.x * inv);
    sM[nl][lane * 4 + 1] = f2bf(acc.y * inv);
    sM[nl][lane * 4 + 2] = f2bf(acc.z * inv);
    sM[nl][lane * 4 + 3] = f2bf(acc.w * inv);
  }
  __syncthreads();

  // Phase 2: dual WMMA GEMM over K=128 with fused epilogue.
  int fbase = wv * 16;
  if (fbase >= dout) return;
  int g = lane >> 4;   // half-wave group
  int m = lane & 15;   // A row / B,D column

  v8f acc = {};
#pragma unroll
  for (int ks = 0; ks < 4; ++ks) {
    int kbase = ks * 32;
    v16bf aM, aX, bL, bR;
#pragma unroll
    for (int i = 0; i < 16; ++i) {
      // 16-bit A-matrix 16x32 layout: lane m holds row M=m;
      // element i lives at K = 16*(i/8) + 8*g + 2*((i/2)%4) + (i%2)
      int ka = kbase + ((i >> 3) << 4) + (g << 3) + (((i >> 1) & 3) << 1) + (i & 1);
      aM[i] = bfbits(sM[m][ka]);
      aX[i] = bfbits(sX[m][ka]);
      // B-matrix 32x16 layout: same K order as A, column = m.
      int kb = kbase + ((i >> 3) << 4) + (g << 3) + (i & 7);
      int f  = fbase + m;
      bL[i] = bfbits(wl[f * IND + kb]);
      bR[i] = bfbits(wr[f * IND + kb]);
    }
    acc = __builtin_amdgcn_wmma_f32_16x16x32_bf16(false, aM, false, bL, (short)0, acc, false, false);
    acc = __builtin_amdgcn_wmma_f32_16x16x32_bf16(false, aX, false, bR, (short)0, acc, false, false);
  }

  int f = fbase + m;
  float bb = bias[f];
  float gmul = 0.f, badd = 0.f, mmean = 0.f;
  if (apply_bn) {
    gmul  = bng[f] * rsqrtf(bnv[f] + BN_EPS);
    badd  = bnb[f];
    mmean = bnm[f];
  }
#pragma unroll
  for (int r = 0; r < 8; ++r) {
    // D layout: VGPR r of lane -> (M = r + 8g, N = m)
    int node = tile * 16 + r + (g << 3);
    float v = acc[r] + bb;
    if (apply_bn) {
      v = (v - mmean) * gmul + badd;
      v = v >= 0.f ? v : 0.1f * v;    // leaky_relu(0.1)
    }
    y[(size_t)node * dout + f] = v;
  }
}

// One wave per node: L2-normalize 64 features.
__global__ void l2norm_k(float* __restrict__ y, int n) {
  int wave = (blockIdx.x * blockDim.x + threadIdx.x) >> 5;
  int lane = threadIdx.x & 31;
  if (wave >= n) return;
  float* p = y + (size_t)wave * OUTD;
  float a = p[lane];
  float b = p[lane + 32];
  float s = a * a + b * b;
#pragma unroll
  for (int off = 16; off > 0; off >>= 1)
    s += __shfl_xor(s, off, 32);
  float nrm = fmaxf(sqrtf(s), 1e-12f);
  p[lane]      = a / nrm;
  p[lane + 32] = b / nrm;
}

extern "C" void kernel_launch(void* const* d_in, const int* in_sizes, int n_in,
                              void* d_out, int out_size, void* d_ws, size_t ws_size,
                              hipStream_t stream) {
  (void)in_sizes; (void)n_in; (void)out_size; (void)ws_size;
  const float* x    = (const float*)d_in[0];
  const int*   ei   = (const int*)d_in[1];     // int64 in source demotes to int32 (no jax x64)
  const float* ew   = (const float*)d_in[2];
  const float* W1l  = (const float*)d_in[3];
  const float* b1   = (const float*)d_in[4];
  const float* W1r  = (const float*)d_in[5];
  const float* W2l  = (const float*)d_in[6];
  const float* b2   = (const float*)d_in[7];
  const float* W2r  = (const float*)d_in[8];
  const float* W3l  = (const float*)d_in[9];
  const float* b3   = (const float*)d_in[10];
  const float* W3r  = (const float*)d_in[11];
  const float* bn1g = (const float*)d_in[12];
  const float* bn1b = (const float*)d_in[13];
  const float* bn1m = (const float*)d_in[14];
  const float* bn1v = (const float*)d_in[15];
  const float* bn2g = (const float*)d_in[16];
  const float* bn2b = (const float*)d_in[17];
  const float* bn2m = (const float*)d_in[18];
  const float* bn2v = (const float*)d_in[19];

  const int* srcI = ei;        // edge_index[0]
  const int* dstI = ei + NE;   // edge_index[1]

  char* ws = (char*)d_ws;
  size_t off = 0;
  float* hA   = (float*)(ws + off); off += (size_t)NN * IND * sizeof(float);
  float* hB   = (float*)(ws + off); off += (size_t)NN * IND * sizeof(float);
  int*   row    = (int*)(ws + off); off += (size_t)(NN + 1) * sizeof(int);
  int*   cursor = (int*)(ws + off); off += (size_t)NN * sizeof(int);
  int*   esrc   = (int*)(ws + off); off += (size_t)NE * sizeof(int);
  float* eww    = (float*)(ws + off); off += (size_t)NE * sizeof(float);
  unsigned short* w1lb = (unsigned short*)(ws + off); off += 128 * 128 * 2;
  unsigned short* w1rb = (unsigned short*)(ws + off); off += 128 * 128 * 2;
  unsigned short* w2lb = (unsigned short*)(ws + off); off += 128 * 128 * 2;
  unsigned short* w2rb = (unsigned short*)(ws + off); off += 128 * 128 * 2;
  unsigned short* w3lb = (unsigned short*)(ws + off); off += 64 * 128 * 2;
  unsigned short* w3rb = (unsigned short*)(ws + off); off += 64 * 128 * 2;

  const int node_blocks = (NN + 255) / 256;
  const int edge_blocks = (NE + 255) / 256;
  const int tile_blocks = (NN + 15) / 16; // 16-node tiles

  // ---- CSR build (once per launch) ----
  fill_zero_i32<<<node_blocks, 256, 0, stream>>>(cursor, NN);
  degree_k<<<edge_blocks, 256, 0, stream>>>(dstI, cursor, NE);
  exscan_k<<<1, 256, 0, stream>>>(cursor, row, NN);
  fill_zero_i32<<<node_blocks, 256, 0, stream>>>(cursor, NN);
  place_k<<<edge_blocks, 256, 0, stream>>>(srcI, dstI, ew, row, cursor, esrc, eww, NE);

  // bf16 weight copies (once per launch; tiny)
  f32_to_bf16_k<<<64, 256, 0, stream>>>(W1l, w1lb, 128 * 128);
  f32_to_bf16_k<<<64, 256, 0, stream>>>(W1r, w1rb, 128 * 128);
  f32_to_bf16_k<<<64, 256, 0, stream>>>(W2l, w2lb, 128 * 128);
  f32_to_bf16_k<<<64, 256, 0, stream>>>(W2r, w2rb, 128 * 128);
  f32_to_bf16_k<<<32, 256, 0, stream>>>(W3l, w3lb, 64 * 128);
  f32_to_bf16_k<<<32, 256, 0, stream>>>(W3r, w3rb, 64 * 128);

  // ---- layer 1: x -> hA ----
  sage_fused_k<<<tile_blocks, 256, 0, stream>>>(x, row, esrc, eww, w1lb, w1rb, b1,
                                                bn1g, bn1b, bn1m, bn1v, hA, 128, 1);
  // ---- layer 2: hA -> hB ----
  sage_fused_k<<<tile_blocks, 256, 0, stream>>>(hA, row, esrc, eww, w2lb, w2rb, b2,
                                                bn2g, bn2b, bn2m, bn2v, hB, 128, 1);
  // ---- layer 3: hB -> out (no BN/ReLU) ----
  sage_fused_k<<<tile_blocks, 256, 0, stream>>>(hB, row, esrc, eww, w3lb, w3rb, b3,
                                                bn1g, bn1b, bn1m, bn1v, (float*)d_out, OUTD, 0);
  // ---- row-wise L2 normalize ----
  l2norm_k<<<(NN + 7) / 8, 256, 0, stream>>>((float*)d_out, NN);
}
